// LAINRDecoder_53085795779231
// MI455X (gfx1250) — matrix operand order repaired and checked by reading, not compiled, not run on hardware
//
#include <hip/hip_runtime.h>
#include <hip/hip_bf16.h>

// ---------------------------------------------------------------------------
// LAINR decoder for MI455X (gfx1250, wave32, WMMA).
// All heavy math runs through v_wmma_f32_16x16x32_bf16 with f32 accumulation.
// Global->LDS staging uses GLOBAL_LOAD_ASYNC_TO_LDS_B128 (ASYNCcnt) when the
// toolchain declares the builtin; falls back to load+ds_store otherwise.
// ---------------------------------------------------------------------------

typedef __attribute__((ext_vector_type(16))) __bf16 v16bf;
typedef __attribute__((ext_vector_type(8)))  float  f32x8;
typedef unsigned short u16;   // raw bf16 bits

__device__ __forceinline__ float b2f(u16 h) {
    unsigned u = ((unsigned)h) << 16;
    float f; __builtin_memcpy(&f, &u, 4); return f;
}
__device__ __forceinline__ u16 f2b(float f) {
    unsigned u; __builtin_memcpy(&u, &f, 4);
    unsigned r = u + 0x7FFFu + ((u >> 16) & 1u);     // round-to-nearest-even
    return (u16)(r >> 16);
}
__device__ __forceinline__ f32x8 zero8() {
    f32x8 z;
#pragma unroll
    for (int i = 0; i < 8; ++i) z[i] = 0.0f;
    return z;
}
// Build a 16x-bf16 fragment from two 16-byte chunks (ISA A/B VGPR layouts).
__device__ __forceinline__ v16bf load_frag2(const u16* p0, const u16* p1) {
    union { uint4 q[2]; v16bf v; } u;
    u.q[0] = *(const uint4*)p0;
    u.q[1] = *(const uint4*)p1;
    return u.v;
}

// ---------------------------------------------------------------------------
// Async global->LDS copy (CDNA5 GLOBAL_LOAD_ASYNC_TO_LDS_B128, ASYNCcnt).
// Builtin signature (from hipcc diagnostic): pointers to 16-byte int vectors,
// global src in AS1, LDS dst in AS3. Generic LDS pointers carry the byte
// offset in their low 32 bits (ISA 10.2), so inttoptr
// casts are sufficient and avoid addrspace-cast diagnostics.
// ---------------------------------------------------------------------------
#if __has_builtin(__builtin_amdgcn_global_load_async_to_lds_b128)
#define HAVE_ASYNC_LDS 1
#else
#define HAVE_ASYNC_LDS 0
#endif

typedef int v4i __attribute__((vector_size(16)));
typedef __attribute__((address_space(1))) v4i gv4i;
typedef __attribute__((address_space(3))) v4i lv4i;

__device__ __forceinline__ void copy16_to_lds(void* lds_dst, const void* gsrc) {
#if HAVE_ASYNC_LDS
    gv4i* g = (gv4i*)(unsigned long long)(__SIZE_TYPE__)gsrc;
    lv4i* l = (lv4i*)(unsigned int)(unsigned long long)(__SIZE_TYPE__)lds_dst;
    __builtin_amdgcn_global_load_async_to_lds_b128(g, l, 0, 0);
#else
    *(uint4*)lds_dst = *(const uint4*)gsrc;
#endif
}
__device__ __forceinline__ void async_copy_wait() {
#if HAVE_ASYNC_LDS
#if __has_builtin(__builtin_amdgcn_s_wait_asynccnt)
    __builtin_amdgcn_s_wait_asynccnt(0);
#else
    asm volatile("s_wait_asynccnt 0x0" ::: "memory");
#endif
#endif
}

// ---------------------------------------------------------------------------
// f32 -> bf16 conversion
// ---------------------------------------------------------------------------
__global__ __launch_bounds__(256) void conv_bf16_kernel(const float* __restrict__ src,
                                                        u16* __restrict__ dst, int n) {
    int i = blockIdx.x * 256 + threadIdx.x;
    if (i < n) dst[i] = f2b(src[i]);
}

// ---------------------------------------------------------------------------
// bf16 elementwise add: dst = a + b
// ---------------------------------------------------------------------------
__global__ __launch_bounds__(256) void add_bf16_kernel(const u16* __restrict__ a,
                                                       const u16* __restrict__ b,
                                                       u16* __restrict__ dst, size_t n) {
    size_t i = (size_t)blockIdx.x * 256 + threadIdx.x;
    if (i < n) dst[i] = f2b(b2f(a[i]) + b2f(b[i]));
}

// ---------------------------------------------------------------------------
// Fourier features (4 sigma bands) + ALiBi "t" vector.
// gamma layout per coord dim d: [d*64 + j] = sin, [d*64 + 32 + j] = cos.
// ---------------------------------------------------------------------------
__global__ __launch_bounds__(256) void prep_kernel(const float* __restrict__ x,
                                                   float* __restrict__ tvec,
                                                   u16* __restrict__ gQ,
                                                   u16* __restrict__ g0,
                                                   u16* __restrict__ g1,
                                                   u16* __restrict__ g2) {
    int hw = blockIdx.x * 256 + threadIdx.x;
    if (hw >= 16384) return;
    float gx = x[hw * 2 + 0];
    float gy = x[hw * 2 + 1];
    int row = (int)(gx * 16.0f);
    int col = (int)(gy * 16.0f);
    tvec[hw] = (float)(row * 16 + col) * (1.0f / 256.0f);

    u16* gs[4]  = { gQ, g0, g1, g2 };
    // log10 of {128, 16, 64, 256}
    const float l10[4] = { 2.10720997f, 1.20411998f, 1.80617997f, 2.40823997f };
    const float PI = 3.14159265358979f;
    const float LN10 = 2.30258509299405f;
#pragma unroll
    for (int s = 0; s < 4; ++s) {
        u16* g = gs[s] + (size_t)hw * 128;
        float step = (l10[s] - 1.0f) * (1.0f / 31.0f);
        for (int j = 0; j < 32; ++j) {
            float om = __expf((1.0f + j * step) * LN10);   // 10^(1 + j*step)
            float a0 = PI * gx * om;
            float a1 = PI * gy * om;
            g[j]       = f2b(__sinf(a0));
            g[32 + j]  = f2b(__cosf(a0));
            g[64 + j]  = f2b(__sinf(a1));
            g[96 + j]  = f2b(__cosf(a1));
        }
    }
}

// ---------------------------------------------------------------------------
// Generic bf16 GEMM:  C[M,N] = op( A[M,K] * B[K,N] [+ bias[N]] [+ addRow] )
// addRow residual is broadcast over batch: addRow[mrow & 16383, N].
// Tile: 128x128 per workgroup (256 thr = 8 waves, per-wave 32x64), K-step 32.
// WMMA: v_wmma_f32_16x16x32_bf16, fragment layouts per CDNA5 ISA 7.12.2.
// ---------------------------------------------------------------------------
template <bool HASBIAS, bool HASADD, bool RELU>
__global__ __launch_bounds__(256) void gemm_bf16_kernel(
    const u16* __restrict__ A, const u16* __restrict__ B,
    const float* __restrict__ bias,
    const u16* __restrict__ addRow,
    u16* __restrict__ C, int M, int N, int K) {

    __shared__ u16 As[128 * 32];    // A tile, row-major [128][32]
    __shared__ u16 Bts[128 * 32];   // B tile transposed: [n][k]

    const int tid  = threadIdx.x;
    const int lane = tid & 31;
    const int w    = tid >> 5;
    const int wm   = w >> 1;         // 0..3 (M direction, 32 rows each)
    const int wn   = w & 1;          // 0..1 (N direction, 64 cols each)
    const int lm   = lane & 15;
    const int hi   = lane >> 4;
    const int m0   = blockIdx.y * 128;
    const int n0   = blockIdx.x * 128;

    f32x8 acc[2][4];
#pragma unroll
    for (int fr = 0; fr < 2; ++fr)
#pragma unroll
        for (int fc = 0; fc < 4; ++fc) acc[fr][fc] = zero8();

    // Per-thread staging coordinates (fixed across the K loop).
    const int ar  = tid >> 2;            // A rows  0..63  (x2 halves below)
    const int akc = tid & 3;             // A 8-half chunk within 32
    const int bk  = tid >> 4;            // B k rows 0..15 (x2 below)
    const int bnc = tid & 15;            // B 8-half chunk within 128

    for (int k0 = 0; k0 < K; k0 += 32) {
        // --- stage A tile via async global->LDS (512 16B chunks, 2/thread)
#pragma unroll
        for (int i = 0; i < 2; ++i) {
            int r = ar + i * 64;
            copy16_to_lds(&As[r * 32 + akc * 8],
                          &A[(size_t)(m0 + r) * K + k0 + akc * 8]);
        }
        // --- stage B tile transposed: B[k0+k][n0+n] -> Bts[n][k]
#pragma unroll
        for (int i = 0; i < 2; ++i) {
            int k = bk + i * 16;
            uint4 d = *(const uint4*)&B[(size_t)(k0 + k) * N + n0 + bnc * 8];
            const u16* hp = (const u16*)&d;
#pragma unroll
            for (int j = 0; j < 8; ++j) Bts[(bnc * 8 + j) * 32 + k] = hp[j];
        }
        // prefetch next K tile while this one is consumed
        if (k0 + 32 < K) {
            __builtin_prefetch(&A[(size_t)(m0 + ar) * K + k0 + 32], 0, 0);
            __builtin_prefetch(&B[(size_t)(k0 + 32 + bk) * N + n0 + bnc * 8], 0, 0);
        }
        async_copy_wait();
        __syncthreads();

        v16bf aF[2];
#pragma unroll
        for (int fr = 0; fr < 2; ++fr) {
            int m = wm * 32 + fr * 16 + lm;
            const u16* p = &As[m * 32 + hi * 8];
            aF[fr] = load_frag2(p, p + 16);   // K: {hi*8..+7, 16+hi*8..+7}
        }
#pragma unroll
        for (int fc = 0; fc < 4; ++fc) {
            int n = wn * 64 + fc * 16 + lm;
            const u16* p = &Bts[n * 32 + hi * 8];
            v16bf bF = load_frag2(p, p + 16); // K: {hi*8..+7, 16+hi*8..+7}
#pragma unroll
            for (int fr = 0; fr < 2; ++fr)
                acc[fr][fc] = __builtin_amdgcn_wmma_f32_16x16x32_bf16(
                    false, aF[fr], false, bF, (short)0, acc[fr][fc], false, false);
        }
        __syncthreads();
    }

    // --- epilogue: bias + broadcast-row residual + relu, store bf16
#pragma unroll
    for (int fr = 0; fr < 2; ++fr)
#pragma unroll
        for (int fc = 0; fc < 4; ++fc)
#pragma unroll
            for (int r = 0; r < 8; ++r) {
                int mrow = m0 + wm * 32 + fr * 16 + hi * 8 + r;
                int ncol = n0 + wn * 64 + fc * 16 + lm;
                float v = acc[fr][fc][r];
                if (HASBIAS) v += bias[ncol];
                if (HASADD)  v += b2f(addRow[(size_t)(mrow & 16383) * N + ncol]);
                if (RELU)    v = fmaxf(v, 0.0f);
                C[(size_t)mrow * N + ncol] = f2b(v);
            }
}

// ---------------------------------------------------------------------------
// Shared-token cross attention, one (b, head, 128-query tile) per workgroup.
// scores = Q*K^T * 1/8 + alibi-bias; softmax over 256 keys; out = P*V / sum.
// Dynamic LDS: P(64K) overlays Q(16K)+K(32K); V at +64K; t/max/sum buffers.
// ---------------------------------------------------------------------------
__global__ __launch_bounds__(256) void attn_kernel(const u16* __restrict__ Q,
                                                   const u16* __restrict__ KV,
                                                   const float* __restrict__ tvec,
                                                   u16* __restrict__ OUT) {
    extern __shared__ char smem[];
    u16*   Qsh  = (u16*)(smem);             // [128][64]
    u16*   Ksh  = (u16*)(smem + 16384);     // [256][64]
    u16*   Psh  = (u16*)(smem);             // [128][256]  (overlays Q,K)
    u16*   Vsh  = (u16*)(smem + 65536);     // [64][256]   (transposed V)
    float* tsh  = (float*)(smem + 98304);   // [128]
    float* maxb = (float*)(smem + 98816);   // [2][128]
    float* sumb = (float*)(smem + 99840);   // [2][128]

    const int tid  = threadIdx.x;
    const int lane = tid & 31;
    const int w    = tid >> 5;
    const int wm   = w >> 1;     // 0..3
    const int wn   = w & 1;      // 0..1
    const int lm   = lane & 15;
    const int hi   = lane >> 4;
    const int qt   = blockIdx.x;     // query tile (HW/128)
    const int h    = blockIdx.y;     // head
    const int b    = blockIdx.z;     // batch

    // ---- phase 0: load K (async), Q (async), V (transpose), t-vector
    {
        int tok = tid;   // 256 threads == 256 tokens
        const u16* kr = &KV[((size_t)(b * 256 + tok)) * 256 + h * 64];
#pragma unroll
        for (int c = 0; c < 8; ++c)
            copy16_to_lds(&Ksh[tok * 64 + c * 8], &kr[c * 8]);
        int r = tid >> 1, half = tid & 1;
        const u16* qr = &Q[((size_t)(qt * 128 + r)) * 128 + h * 64 + half * 32];
#pragma unroll
        for (int c = 0; c < 4; ++c)
            copy16_to_lds(&Qsh[r * 64 + half * 32 + c * 8], &qr[c * 8]);
        const u16* vr = &KV[((size_t)(b * 256 + tok)) * 256 + 128 + h * 64];
#pragma unroll
        for (int c = 0; c < 8; ++c) {
            uint4 d = *(const uint4*)&vr[c * 8];
            const u16* hp = (const u16*)&d;
#pragma unroll
            for (int j = 0; j < 8; ++j) Vsh[(c * 8 + j) * 256 + tok] = hp[j];
        }
        if (tid < 128) tsh[tid] = tvec[qt * 128 + tid];
    }
    async_copy_wait();
    __syncthreads();

    // ---- phase 1: S = Q*K^T  (per-wave 32 rows x 128 tokens, 16 frags)
    f32x8 sA[2][8];
#pragma unroll
    for (int fr = 0; fr < 2; ++fr)
#pragma unroll
        for (int fc = 0; fc < 8; ++fc) sA[fr][fc] = zero8();

#pragma unroll
    for (int kk = 0; kk < 64; kk += 32) {
        v16bf aF[2];
#pragma unroll
        for (int fr = 0; fr < 2; ++fr) {
            int m = wm * 32 + fr * 16 + lm;
            const u16* p = &Qsh[m * 64 + kk + hi * 8];
            aF[fr] = load_frag2(p, p + 16);
        }
#pragma unroll
        for (int fc = 0; fc < 8; ++fc) {
            int n = wn * 128 + fc * 16 + lm;
            const u16* p = &Ksh[n * 64 + kk + hi * 8];
            v16bf bF = load_frag2(p, p + 16);
#pragma unroll
            for (int fr = 0; fr < 2; ++fr)
                sA[fr][fc] = __builtin_amdgcn_wmma_f32_16x16x32_bf16(
                    false, aF[fr], false, bF, (short)0, sA[fr][fc], false, false);
        }
    }

    // scale + ALiBi-like bias
    const float scale = 0.125f;   // 64^-0.5
#pragma unroll
    for (int fr = 0; fr < 2; ++fr)
#pragma unroll
        for (int fc = 0; fc < 8; ++fc)
#pragma unroll
            for (int r = 0; r < 8; ++r) {
                int rl = wm * 32 + fr * 16 + hi * 8 + r;
                int n  = wn * 128 + fc * 16 + lm;
                float dt = tsh[rl] - (n + 0.5f) * (1.0f / 256.0f);
                sA[fr][fc][r] = sA[fr][fc][r] * scale - 10.0f * dt * dt;
            }

    // ---- softmax over 256 keys
#pragma unroll
    for (int fr = 0; fr < 2; ++fr)
#pragma unroll
        for (int r = 0; r < 8; ++r) {
            float mx = -1e30f;
#pragma unroll
            for (int fc = 0; fc < 8; ++fc) mx = fmaxf(mx, sA[fr][fc][r]);
#pragma unroll
            for (int d = 1; d < 16; d <<= 1) mx = fmaxf(mx, __shfl_xor(mx, d, 32));
            int rl = wm * 32 + fr * 16 + hi * 8 + r;
            if (lm == 0) maxb[wn * 128 + rl] = mx;
        }
    __syncthreads();
#pragma unroll
    for (int fr = 0; fr < 2; ++fr)
#pragma unroll
        for (int r = 0; r < 8; ++r) {
            int rl = wm * 32 + fr * 16 + hi * 8 + r;
            float gm = fmaxf(maxb[rl], maxb[128 + rl]);
            float s = 0.0f;
#pragma unroll
            for (int fc = 0; fc < 8; ++fc) {
                float p = __expf(sA[fr][fc][r] - gm);
                sA[fr][fc][r] = p;
                s += p;
            }
#pragma unroll
            for (int d = 1; d < 16; d <<= 1) s += __shfl_xor(s, d, 32);
            if (lm == 0) sumb[wn * 128 + rl] = s;
        }
    __syncthreads();   // also guarantees all Q/K LDS reads done before P writes

    // P (bf16) -> LDS in A layout
#pragma unroll
    for (int fr = 0; fr < 2; ++fr)
#pragma unroll
        for (int fc = 0; fc < 8; ++fc)
#pragma unroll
            for (int r = 0; r < 8; ++r) {
                int rl = wm * 32 + fr * 16 + hi * 8 + r;
                int n  = wn * 128 + fc * 16 + lm;
                Psh[rl * 256 + n] = f2b(sA[fr][fc][r]);
            }
    __syncthreads();

    // ---- phase 2: O = P*V  (per-wave 32 rows x 32 dims)
    f32x8 oA[2][2];
#pragma unroll
    for (int fr = 0; fr < 2; ++fr)
#pragma unroll
        for (int fc = 0; fc < 2; ++fc) oA[fr][fc] = zero8();

    for (int kk = 0; kk < 256; kk += 32) {
        v16bf aF[2];
#pragma unroll
        for (int fr = 0; fr < 2; ++fr) {
            int m = wm * 32 + fr * 16 + lm;
            const u16* p = &Psh[m * 256 + kk + hi * 8];
            aF[fr] = load_frag2(p, p + 16);
        }
#pragma unroll
        for (int fc = 0; fc < 2; ++fc) {
            int n = wn * 32 + fc * 16 + lm;
            const u16* p = &Vsh[n * 256 + kk + hi * 8];
            v16bf bF = load_frag2(p, p + 16);
#pragma unroll
            for (int fr = 0; fr < 2; ++fr)
                oA[fr][fc] = __builtin_amdgcn_wmma_f32_16x16x32_bf16(
                    false, aF[fr], false, bF, (short)0, oA[fr][fc], false, false);
        }
    }

    // normalize and store
#pragma unroll
    for (int fr = 0; fr < 2; ++fr)
#pragma unroll
        for (int fc = 0; fc < 2; ++fc)
#pragma unroll
            for (int r = 0; r < 8; ++r) {
                int rl = wm * 32 + fr * 16 + hi * 8 + r;
                int n  = wn * 32 + fc * 16 + lm;
                float rs = sumb[rl] + sumb[128 + rl];
                float v = oA[fr][fc][r] / rs;
                size_t row = (size_t)b * 16384 + (size_t)qt * 128 + rl;
                OUT[row * 128 + h * 64 + n] = f2b(v);
            }
}

// ---------------------------------------------------------------------------
// total[r,c] = sum_i  hv_i[r,:] . out_W[i][:,c]  + sum_i out_b[i][c]
// ---------------------------------------------------------------------------
__global__ __launch_bounds__(256) void finalize_kernel(const u16* __restrict__ h0,
                                                       const u16* __restrict__ h1,
                                                       const u16* __restrict__ h2,
                                                       const float* __restrict__ oW,
                                                       const float* __restrict__ ob,
                                                       float* __restrict__ out, int rows) {
    int r = blockIdx.x * 256 + threadIdx.x;
    if (r >= rows) return;
    float acc0 = ob[0] + ob[3] + ob[6];
    float acc1 = ob[1] + ob[4] + ob[7];
    float acc2 = ob[2] + ob[5] + ob[8];
    const u16* hb[3] = { h0, h1, h2 };
    for (int band = 0; band < 3; ++band) {
        const u16* hr = hb[band] + (size_t)r * 512;
        const float* wb = oW + band * 1536;
        for (int kk = 0; kk < 64; ++kk) {
            uint4 d = *(const uint4*)&hr[kk * 8];
            const u16* hp = (const u16*)&d;
#pragma unroll
            for (int e = 0; e < 8; ++e) {
                float v = b2f(hp[e]);
                int k = kk * 8 + e;
                acc0 += v * wb[k * 3 + 0];
                acc1 += v * wb[k * 3 + 1];
                acc2 += v * wb[k * 3 + 2];
            }
        }
    }
    out[(size_t)r * 3 + 0] = acc0;
    out[(size_t)r * 3 + 1] = acc1;
    out[(size_t)r * 3 + 2] = acc2;
}

// ---------------------------------------------------------------------------
// Host orchestration
// ---------------------------------------------------------------------------
extern "C" void kernel_launch(void* const* d_in, const int* in_sizes, int n_in,
                              void* d_out, int out_size, void* d_ws, size_t ws_size,
                              hipStream_t stream) {
    (void)in_sizes; (void)n_in; (void)out_size; (void)ws_size;

    const float* x      = (const float*)d_in[0];   // [8,16384,2]
    const float* tokens = (const float*)d_in[1];   // [8,256,512]
    const float* q_W    = (const float*)d_in[2];   // [128,512]
    const float* q_b    = (const float*)d_in[3];   // [512]
    const float* Wq     = (const float*)d_in[4];   // [512,128]
    const float* Wkv    = (const float*)d_in[5];   // [512,256]
    const float* Wo     = (const float*)d_in[6];   // [128,512]
    const float* bo     = (const float*)d_in[7];   // [512]
    const float* bw_W   = (const float*)d_in[8];   // [3,128,512]
    const float* bw_b   = (const float*)d_in[9];   // [3,512]
    const float* mod_W  = (const float*)d_in[10];  // [3,512,512]
    const float* mod_b  = (const float*)d_in[11];  // [3,512]
    const float* hv_W   = (const float*)d_in[12];  // [2,512,512]
    const float* hv_b   = (const float*)d_in[13];  // [2,512]
    const float* out_W  = (const float*)d_in[14];  // [3,512,3]
    const float* out_b  = (const float*)d_in[15];  // [3,3]
    float* out          = (float*)d_out;           // [8,16384,3]

    const int HW = 16384, HID = 512, M_TOK = 256, BATCH = 8;
    const int BHW = BATCH * HW;                    // 131072

    char* ws = (char*)d_ws;
    size_t off = 0;
    auto alloc = [&](size_t bytes) -> size_t {
        size_t o = off;
        off = (off + bytes + 255) & ~(size_t)255;
        return o;
    };

    // bf16 weights / tokens
    size_t o_qW  = alloc((size_t)128 * 512 * 2);
    size_t o_Wq  = alloc((size_t)512 * 128 * 2);
    size_t o_Wkv = alloc((size_t)512 * 256 * 2);
    size_t o_Wo  = alloc((size_t)128 * 512 * 2);
    size_t o_bw  = alloc((size_t)3 * 128 * 512 * 2);
    size_t o_mod = alloc((size_t)3 * 512 * 512 * 2);
    size_t o_hv  = alloc((size_t)2 * 512 * 512 * 2);
    size_t o_tok = alloc((size_t)BATCH * M_TOK * HID * 2);
    // activations
    size_t o_tv  = alloc((size_t)HW * 4);
    size_t o_gQ  = alloc((size_t)HW * 128 * 2);
    size_t o_g0  = alloc((size_t)HW * 128 * 2);
    size_t o_g1  = alloc((size_t)HW * 128 * 2);
    size_t o_g2  = alloc((size_t)HW * 128 * 2);
    size_t o_XQ  = alloc((size_t)HW * HID * 2);
    size_t o_Qm  = alloc((size_t)HW * 128 * 2);
    size_t o_KV  = alloc((size_t)BATCH * M_TOK * 256 * 2);
    size_t o_OUT = alloc((size_t)BHW * 128 * 2);
    size_t o_HL  = alloc((size_t)3 * HW * HID * 2);
    size_t o_MOD = alloc((size_t)BHW * HID * 2);
    size_t o_M0  = alloc((size_t)BHW * HID * 2);
    size_t o_T1  = alloc((size_t)BHW * HID * 2);
    size_t o_T2  = alloc((size_t)BHW * HID * 2);

    u16* w_qW  = (u16*)(ws + o_qW);
    u16* w_Wq  = (u16*)(ws + o_Wq);
    u16* w_Wkv = (u16*)(ws + o_Wkv);
    u16* w_Wo  = (u16*)(ws + o_Wo);
    u16* w_bw  = (u16*)(ws + o_bw);
    u16* w_mod = (u16*)(ws + o_mod);
    u16* w_hv  = (u16*)(ws + o_hv);
    u16* w_tok = (u16*)(ws + o_tok);
    float* tv  = (float*)(ws + o_tv);
    u16* gQ    = (u16*)(ws + o_gQ);
    u16* g0    = (u16*)(ws + o_g0);
    u16* g1    = (u16*)(ws + o_g1);
    u16* g2    = (u16*)(ws + o_g2);
    u16* XQ    = (u16*)(ws + o_XQ);
    u16* Qm    = (u16*)(ws + o_Qm);
    u16* KVm   = (u16*)(ws + o_KV);
    u16* OUTa  = (u16*)(ws + o_OUT);
    u16* HL    = (u16*)(ws + o_HL);
    u16* MOD   = (u16*)(ws + o_MOD);
    u16* M0b   = (u16*)(ws + o_M0);
    u16* T1    = (u16*)(ws + o_T1);
    u16* T2    = (u16*)(ws + o_T2);

    auto conv = [&](const float* s, u16* d, int n) {
        conv_bf16_kernel<<<(n + 255) / 256, 256, 0, stream>>>(s, d, n);
    };
    conv(q_W,    w_qW,  128 * 512);
    conv(Wq,     w_Wq,  512 * 128);
    conv(Wkv,    w_Wkv, 512 * 256);
    conv(Wo,     w_Wo,  128 * 512);
    conv(bw_W,   w_bw,  3 * 128 * 512);
    conv(mod_W,  w_mod, 3 * 512 * 512);
    conv(hv_W,   w_hv,  2 * 512 * 512);
    conv(tokens, w_tok, BATCH * M_TOK * HID);

    // Fourier features + t-vector
    prep_kernel<<<HW / 256, 256, 0, stream>>>(x, tv, gQ, g0, g1, g2);

    // dispatcher over the four used epilogue variants
    auto gemm = [&](const u16* A, const u16* B, const float* bias,
                    const u16* addRow, u16* C, int M, int N, int K, int relu) {
        dim3 grid(N / 128, M / 128);
        if (addRow)
            gemm_bf16_kernel<true, true, true><<<grid, 256, 0, stream>>>(
                A, B, bias, addRow, C, M, N, K);
        else if (bias && relu)
            gemm_bf16_kernel<true, false, true><<<grid, 256, 0, stream>>>(
                A, B, bias, addRow, C, M, N, K);
        else if (bias)
            gemm_bf16_kernel<true, false, false><<<grid, 256, 0, stream>>>(
                A, B, bias, addRow, C, M, N, K);
        else
            gemm_bf16_kernel<false, false, false><<<grid, 256, 0, stream>>>(
                A, B, bias, addRow, C, M, N, K);
    };

    // xq = relu(gammaQ @ q_W + q_b)            [16384,512]
    gemm(gQ, w_qW, q_b, nullptr, XQ, HW, HID, 128, 1);
    // Q = xq @ Wq                              [16384,128]
    gemm(XQ, w_Wq, nullptr, nullptr, Qm, HW, 128, HID, 0);
    // KV = tokens @ Wkv                        [2048,256]
    gemm(w_tok, w_Wkv, nullptr, nullptr, KVm, BATCH * M_TOK, 256, HID, 0);
    // h_l[i] = relu(gamma_l[i] @ bw_W[i] + bw_b[i])   [16384,512] x3
    {
        const u16* gl[3] = { g0, g1, g2 };
        for (int i = 0; i < 3; ++i)
            gemm(gl[i], w_bw + (size_t)i * 128 * 512, bw_b + i * 512,
                 nullptr, HL + (size_t)i * HW * HID, HW, HID, 128, 1);
    }

    // attention -> OUTa [131072,128]
    {
        dim3 grid(HW / 128, 2, BATCH);
        size_t shbytes = 100864;
        attn_kernel<<<grid, 256, shbytes, stream>>>(Qm, KVm, tv, OUTa);
    }

    // MOD = OUTa @ Wo + bo                     [131072,512]
    gemm(OUTa, w_Wo, bo, nullptr, MOD, BHW, HID, 128, 0);

    // band 0: M0 = relu(HL0[hw] + MOD @ mod_W0 + mod_b0)
    gemm(MOD, w_mod + (size_t)0 * 512 * 512, mod_b + 0, HL + (size_t)0 * HW * HID,
         M0b, BHW, HID, HID, 1);
    // band 1: M1 -> T1
    gemm(MOD, w_mod + (size_t)1 * 512 * 512, mod_b + 512, HL + (size_t)1 * HW * HID,
         T1, BHW, HID, HID, 1);
    // T1 = M1 + M0  (sum for hv GEMM)
    {
        size_t n = (size_t)BHW * HID;
        add_bf16_kernel<<<(unsigned)((n + 255) / 256), 256, 0, stream>>>(T1, M0b, T1, n);
    }
    // HV1 = relu(T1 @ hv_W0 + hv_b0) -> T2
    gemm(T1, w_hv + (size_t)0 * 512 * 512, hv_b + 0, nullptr, T2, BHW, HID, HID, 1);
    // band 2: M2 -> T1 (overwrites SUM1, no longer needed)
    gemm(MOD, w_mod + (size_t)2 * 512 * 512, mod_b + 1024, HL + (size_t)2 * HW * HID,
         T1, BHW, HID, HID, 1);
    // T1 = M2 + HV1
    {
        size_t n = (size_t)BHW * HID;
        add_bf16_kernel<<<(unsigned)((n + 255) / 256), 256, 0, stream>>>(T1, T2, T1, n);
    }
    // HV2 = relu(T1 @ hv_W1 + hv_b1) -> MOD (reuse, MOD dead now)
    gemm(T1, w_hv + (size_t)1 * 512 * 512, hv_b + 512, nullptr, MOD, BHW, HID, HID, 1);

    // total = M0 @ oW0 + HV1 @ oW1 + HV2 @ oW2 + sum(out_b)
    finalize_kernel<<<BHW / 256, 256, 0, stream>>>(M0b, T2, MOD, out_W, out_b, out, BHW);
}
